// ZEnhancedDecoderBlock_31086973289058
// MI455X (gfx1250) — matrix operand order repaired and checked by reading, not compiled, not run on hardware
//
#include <hip/hip_runtime.h>
#include <hip/hip_bf16.h>

// ---------------------------------------------------------------------------
// ZEnhancedDecoderBlock for MI455X (gfx1250, wave32, WMMA + TDM)
//
// out[ch,p] = x[ch,p] + gamma*o[z(ch),c(ch)] + rdc * lrelu(fused[ch,p])
//   fused = conv1x1([x; f0; f1; f2], fuse_w)  -> GEMM 16384 x 2096 x 2048
//
// Data plan:
//   XT   : bf16 features, TRANSPOSED + K-padded  [NPIX][KP=2112]
//          cols 0..2047 = x (bf16), 2048..2095 = f0/f1/f2, 2096..2111 = 0
//   Wfrag: weights pre-swizzled into WMMA A-fragment order (bf16), zero-padded
//   B tiles stream into LDS via tensor_load_to_lds (TDM), double buffered.
// ---------------------------------------------------------------------------

typedef __attribute__((ext_vector_type(16))) __bf16 v16bf;
typedef __attribute__((ext_vector_type(8)))  __bf16 v8bf;
typedef __attribute__((ext_vector_type(8)))  float  v8f;
typedef __attribute__((ext_vector_type(4)))  unsigned int v4u;
typedef __attribute__((ext_vector_type(8)))  int v8i;
typedef __attribute__((ext_vector_type(4)))  int v4i;

#define NPIX 16384
#define KP   2112      // padded K (66 tiles of 32)
#define NEG_SLOPE 0.2f

__device__ __forceinline__ float lrelu(float v) {
  return v >= 0.f ? v : NEG_SLOPE * v;
}

// ------------------------- TDM 2-D tile load ------------------------------
// Loads a [rows][32] bf16 tile (row stride KP elements) from global into LDS
// as a contiguous [rows][32] block. Descriptor per CDNA5 ISA 08_async_tensor
// section 8 (D# group0/group1; groups 2/3 zero for 2-D).
__device__ __forceinline__ void tdm_load_tile_2d(unsigned int lds_off,
                                                 const void* gptr, int rows) {
  unsigned long long ga = (unsigned long long)(uintptr_t)gptr;
  v4u g0;
  g0[0] = 1u;                                  // count=1, user mode
  g0[1] = lds_off;                             // lds_addr (bytes)
  g0[2] = (unsigned int)ga;                    // global_addr[31:0]
  g0[3] = (unsigned int)(ga >> 32) | (2u << 30);  // global_addr[56:32] | type=2
  v8i g1;
  g1[0] = 0x00010000;                          // data_size=1 -> 2 bytes/elt
  g1[1] = (32 << 16);                          // tensor_dim0 = 32 (lo16)
  g1[2] = (rows << 16);                        // tensor_dim1 = rows (lo16)
  g1[3] = (32 << 16);                          // tile_dim0 = 32
  g1[4] = rows;                                // tile_dim1 = rows, tile_dim2 = 0
  g1[5] = KP;                                  // tensor_dim0_stride (elements)
  g1[6] = 0;
  g1[7] = 0;
  v4i g2 = {0, 0, 0, 0};
  v4i g3 = {0, 0, 0, 0};
#if __has_include(<hip/amd_detail/amd_gfx1250_TDM.h>)
  v8i g4 = {0, 0, 0, 0, 0, 0, 0, 0};
  __builtin_amdgcn_tensor_load_to_lds(g0, g1, g2, g3, g4, 0);
#else
  __builtin_amdgcn_tensor_load_to_lds(g0, g1, g2, g3, 0);
#endif
}

// --------------------------- fragment loaders ------------------------------
// B (32x16, K x N): lane l (n=l&15, half=l>>4) holds element j -> k=16*half+j.
// LDS tile stored [n][k] (row = 32 bf16) -> one contiguous 32B run per lane.
__device__ __forceinline__ v16bf lds_frag_b(const __bf16* Xt, int nLocal, int half) {
  return *(const v16bf*)(Xt + nLocal * 32 + half * 16);
}

// --------------------------- kernel 0: init -------------------------------
__global__ void init_acc_kernel(float* __restrict__ T, float* __restrict__ S) {
  int i = blockIdx.x * 256 + threadIdx.x;
  if (i < 2048) T[i] = 0.f;
  if (i < 32)   S[i] = 0.f;
}

// ------------- kernel 1: fp32 -> bf16 transpose + LN statistics -----------
// grid (NPIX/256, 32); thread owns one pixel of one z-slice.
// XT[p][z*64+c] = bf16(x[z,c,p]);  T[z*64+c] = sum_p rs*x;  S[z] = sum_p rs*mu
__global__ __launch_bounds__(256)
void prep_kernel(const float* __restrict__ x, __bf16* __restrict__ XT,
                 float* __restrict__ T, float* __restrict__ S) {
  const int p = blockIdx.x * 256 + threadIdx.x;
  const int z = blockIdx.y;
  const float* xr = x + (size_t)z * 64 * NPIX + p;
  __bf16* xo = XT + (size_t)p * KP + z * 64;

  float regs[64];
  float sum = 0.f, sumsq = 0.f;
#pragma unroll
  for (int c = 0; c < 64; ++c) {
    float v = xr[(size_t)c * NPIX];
    regs[c] = v;
    sum += v;
    sumsq += v * v;
  }
#pragma unroll
  for (int cc = 0; cc < 8; ++cc) {
    v8bf pk;
#pragma unroll
    for (int e = 0; e < 8; ++e) pk[e] = (__bf16)regs[cc * 8 + e];
    *(v8bf*)(xo + cc * 8) = pk;
  }
  if (z == 0) {  // zero the K padding columns once per pixel
    const __bf16 zb = (__bf16)0.0f;
    v8bf zv = {zb, zb, zb, zb, zb, zb, zb, zb};
    *(v8bf*)(XT + (size_t)p * KP + 2096) = zv;
    *(v8bf*)(XT + (size_t)p * KP + 2104) = zv;
  }

  const float mu  = sum * (1.f / 64.f);
  const float var = sumsq * (1.f / 64.f) - mu * mu;
  const float rs  = rsqrtf(var + 1e-5f);

  __shared__ float accT[64];
  __shared__ float accS;
  if (threadIdx.x < 64) accT[threadIdx.x] = 0.f;
  if (threadIdx.x == 0) accS = 0.f;
  __syncthreads();
  atomicAdd(&accS, rs * mu);
#pragma unroll
  for (int c = 0; c < 64; ++c) atomicAdd(&accT[c], rs * regs[c]);
  __syncthreads();
  if (threadIdx.x < 64) atomicAdd(&T[z * 64 + threadIdx.x], accT[threadIdx.x]);
  if (threadIdx.x == 0) atomicAdd(&S[z], accS);
}

// ------------- kernel 2: weight swizzle into A-fragment order -------------
// A (16x32, M x K): lane l (m=l&15, half=l>>4), element j ->
//   k = 16*half + j.
// Layout: out[((t*(M/16)+mT)*32 + lane)*16 + j], zero-padded past Kk.
__global__ void wprep_kernel(const float* __restrict__ W, int Kk, int M,
                             __bf16* __restrict__ out, int nTiles) {
  const int gid = blockIdx.x * 256 + threadIdx.x;
  const int total = nTiles * (M >> 4) * 32;
  if (gid >= total) return;
  const int lane = gid & 31;
  const int mT   = (gid >> 5) % (M >> 4);
  const int t    = (gid >> 5) / (M >> 4);
  const int m    = mT * 16 + (lane & 15);
  const int half = lane >> 4;
  v16bf pk;
#pragma unroll
  for (int j = 0; j < 16; ++j) {
    const int k = t * 32 + half * 16 + j;
    const float v = (k < Kk) ? W[(size_t)k * M + m] : 0.f;
    pk[j] = (__bf16)v;
  }
  *(v16bf*)(out + (size_t)gid * 16) = pk;
}

// ------------------- kernel 3: tiny z-attention (1 wave) ------------------
__global__ void attn_kernel(const float* __restrict__ T, const float* __restrict__ S,
                            const float* __restrict__ ln_w, const float* __restrict__ ln_b,
                            const float* __restrict__ wq, const float* __restrict__ bq,
                            const float* __restrict__ wk, const float* __restrict__ bk,
                            const float* __restrict__ wv, const float* __restrict__ bv,
                            const float* __restrict__ wo, const float* __restrict__ bo,
                            const float* __restrict__ rel, const float* __restrict__ gamma_p,
                            float* __restrict__ addvec) {
  __shared__ float pooled[32][64];
  __shared__ float qs[32][16], ks[32][16], vs[32][16];
  const int i = threadIdx.x;  // z row, 0..31

  const float Sz = S[i];
  for (int c = 0; c < 64; ++c)
    pooled[i][c] = ln_w[c] * ((T[i * 64 + c] - Sz) * (1.f / (float)NPIX)) + ln_b[c];
  __syncthreads();
  for (int o = 0; o < 16; ++o) {
    float aq = bq[o], ak = bk[o], av = bv[o];
    for (int c = 0; c < 64; ++c) {
      const float pv = pooled[i][c];
      aq += pv * wq[c * 16 + o];
      ak += pv * wk[c * 16 + o];
      av += pv * wv[c * 16 + o];
    }
    qs[i][o] = aq; ks[i][o] = ak; vs[i][o] = av;
  }
  __syncthreads();
  float oo[16];
#pragma unroll
  for (int h = 0; h < 4; ++h) {
    float logits[32];
    float mx = -1e30f;
    for (int j = 0; j < 32; ++j) {
      float d = 0.f;
#pragma unroll
      for (int dd = 0; dd < 4; ++dd) d += qs[i][h * 4 + dd] * ks[j][h * 4 + dd];
      int ridx = j - i + 32;
      ridx = min(max(ridx, 0), 64);
      const float l = d * 0.5f + rel[h * 65 + ridx];
      logits[j] = l;
      mx = fmaxf(mx, l);
    }
    float den = 0.f;
    for (int j = 0; j < 32; ++j) { logits[j] = __expf(logits[j] - mx); den += logits[j]; }
    const float inv = 1.f / den;
#pragma unroll
    for (int dd = 0; dd < 4; ++dd) {
      float s = 0.f;
      for (int j = 0; j < 32; ++j) s += logits[j] * vs[j][h * 4 + dd];
      oo[h * 4 + dd] = s * inv;
    }
  }
  const float gamma = *gamma_p;
  for (int c = 0; c < 64; ++c) {
    float s = bo[c];
#pragma unroll
    for (int o = 0; o < 16; ++o) s += oo[o] * wo[o * 64 + c];
    addvec[i * 64 + c] = gamma * s;
  }
}

// ---------------- kernel 4/5/6: growth convs (M=16, WMMA + TDM) -----------
// Block tile 16 x 128; 8 waves, one 16x16 tile each; writes bf16 transposed
// back into XT at column 2048+fofs.
__global__ __launch_bounds__(256)
void rdb_gemm_kernel(const __bf16* __restrict__ XT, const __bf16* __restrict__ WF,
                     const float* __restrict__ bias, int nTiles,
                     __bf16* __restrict__ XTout, int fofs) {
  constexpr int NT = 128;
  __shared__ __align__(32) __bf16 Xt[2][NT * 32];
  const int tid  = threadIdx.x;
  const int lane = tid & 31;
  const int wave = tid >> 5;
  const int half = lane >> 4;
  const int pbase = blockIdx.x * NT;

  const unsigned int lds0 = (unsigned int)(uintptr_t)&Xt[0][0];
  const unsigned int lds1 = (unsigned int)(uintptr_t)&Xt[1][0];

  v8f acc = {0.f, 0.f, 0.f, 0.f, 0.f, 0.f, 0.f, 0.f};

  if (wave == 0) tdm_load_tile_2d(lds0, XT + (size_t)pbase * KP, NT);

  for (int t = 0; t < nTiles; ++t) {
    if (wave == 0) __builtin_amdgcn_s_wait_tensorcnt(0);
    __syncthreads();
    if (wave == 0 && t + 1 < nTiles)
      tdm_load_tile_2d((t & 1) ? lds0 : lds1,
                       XT + (size_t)pbase * KP + (t + 1) * 32, NT);
    const __bf16* Xc = Xt[t & 1];
    const v16bf a = *(const v16bf*)(WF + ((size_t)t * 32 + lane) * 16);
    const v16bf b = lds_frag_b(Xc, wave * 16 + (lane & 15), half);
    acc = __builtin_amdgcn_wmma_f32_16x16x32_bf16(false, a, false, b,
                                                  (short)0, acc, false, false);
  }

  const int p = pbase + wave * 16 + (lane & 15);
  v8bf pk;
#pragma unroll
  for (int r = 0; r < 8; ++r)
    pk[r] = (__bf16)lrelu(acc[r] + bias[half * 8 + r]);
  *(v8bf*)(XTout + (size_t)p * KP + 2048 + fofs + half * 8) = pk;
}

// ---------------- kernel 7: fused conv + epilogue (WMMA + TDM) ------------
// Block tile 128 (M) x 32 (N); 8 waves over M, each wave 2 N-fragments.
__global__ __launch_bounds__(256)
void fused_gemm_kernel(const __bf16* __restrict__ XT, const __bf16* __restrict__ WF,
                       const float* __restrict__ bias, const float* __restrict__ Xorig,
                       const float* __restrict__ addvec, const float* __restrict__ rdc_p,
                       float* __restrict__ outp) {
  constexpr int NT = 32, NTILES = 66, NMT = 128;  // NMT = 2048/16
  __shared__ __align__(32) __bf16 Xt[2][NT * 32];
  const int tid  = threadIdx.x;
  const int lane = tid & 31;
  const int wave = tid >> 5;
  const int half = lane >> 4;
  const int pbase = blockIdx.x * NT;
  const int mT    = blockIdx.y * 8 + wave;

  const unsigned int lds0 = (unsigned int)(uintptr_t)&Xt[0][0];
  const unsigned int lds1 = (unsigned int)(uintptr_t)&Xt[1][0];

  v8f acc0 = {0.f, 0.f, 0.f, 0.f, 0.f, 0.f, 0.f, 0.f};
  v8f acc1 = {0.f, 0.f, 0.f, 0.f, 0.f, 0.f, 0.f, 0.f};

  if (wave == 0) tdm_load_tile_2d(lds0, XT + (size_t)pbase * KP, NT);

  for (int t = 0; t < NTILES; ++t) {
    if (wave == 0) __builtin_amdgcn_s_wait_tensorcnt(0);
    __syncthreads();
    if (wave == 0 && t + 1 < NTILES)
      tdm_load_tile_2d((t & 1) ? lds0 : lds1,
                       XT + (size_t)pbase * KP + (t + 1) * 32, NT);
    const __bf16* Xc = Xt[t & 1];
    const v16bf a  = *(const v16bf*)(WF + (((size_t)t * NMT + mT) * 32 + lane) * 16);
    const v16bf b0 = lds_frag_b(Xc, lane & 15, half);
    const v16bf b1 = lds_frag_b(Xc, 16 + (lane & 15), half);
    acc0 = __builtin_amdgcn_wmma_f32_16x16x32_bf16(false, a, false, b0,
                                                   (short)0, acc0, false, false);
    acc1 = __builtin_amdgcn_wmma_f32_16x16x32_bf16(false, a, false, b1,
                                                   (short)0, acc1, false, false);
  }

  const float rdc = *rdc_p;
#pragma unroll
  for (int f = 0; f < 2; ++f) {
    const v8f acc = f ? acc1 : acc0;
    const int p = pbase + f * 16 + (lane & 15);
#pragma unroll
    for (int r = 0; r < 8; ++r) {
      const int m = mT * 16 + half * 8 + r;
      const float fused = lrelu(acc[r] + bias[m]);
      const size_t idx = (size_t)m * NPIX + p;
      outp[idx] = Xorig[idx] + addvec[m] + rdc * fused;
    }
  }
}

// ------------------------------- launcher ---------------------------------
extern "C" void kernel_launch(void* const* d_in, const int* in_sizes, int n_in,
                              void* d_out, int out_size, void* d_ws, size_t ws_size,
                              hipStream_t stream) {
  const float* x     = (const float*)d_in[0];
  const float* ln_w  = (const float*)d_in[1];
  const float* ln_b  = (const float*)d_in[2];
  const float* wq    = (const float*)d_in[3];
  const float* bq    = (const float*)d_in[4];
  const float* wk    = (const float*)d_in[5];
  const float* bk    = (const float*)d_in[6];
  const float* wv    = (const float*)d_in[7];
  const float* bv    = (const float*)d_in[8];
  const float* wo    = (const float*)d_in[9];
  const float* bo    = (const float*)d_in[10];
  const float* rel   = (const float*)d_in[11];
  const float* gamma = (const float*)d_in[12];
  const float* dw0   = (const float*)d_in[13];
  const float* db0   = (const float*)d_in[14];
  const float* dw1   = (const float*)d_in[15];
  const float* db1   = (const float*)d_in[16];
  const float* dw2   = (const float*)d_in[17];
  const float* db2   = (const float*)d_in[18];
  const float* fw    = (const float*)d_in[19];
  const float* fb    = (const float*)d_in[20];
  const float* rdc   = (const float*)d_in[21];

  char* ws = (char*)d_ws;
  __bf16* XT  = (__bf16*)ws;                         // [16384][2112] bf16
  size_t  off = (size_t)NPIX * KP * 2;               // 69,206,016
  __bf16* WFf = (__bf16*)(ws + off); off += (size_t)66 * 128 * 32 * 16 * 2;
  __bf16* WF0 = (__bf16*)(ws + off); off += (size_t)66 * 32 * 16 * 2;
  __bf16* WF1 = (__bf16*)(ws + off); off += (size_t)66 * 32 * 16 * 2;
  __bf16* WF2 = (__bf16*)(ws + off); off += (size_t)66 * 32 * 16 * 2;
  float*  T   = (float*)(ws + off);
  float*  S   = T + 2048;
  float*  addvec = S + 32;
  float*  outp = (float*)d_out;

  init_acc_kernel<<<8, 256, 0, stream>>>(T, S);
  prep_kernel<<<dim3(NPIX / 256, 32), 256, 0, stream>>>(x, XT, T, S);
  wprep_kernel<<<1056, 256, 0, stream>>>(fw,  2096, 2048, WFf, 66);
  wprep_kernel<<<8,    256, 0, stream>>>(dw0, 2048, 16,   WF0, 64);
  wprep_kernel<<<9,    256, 0, stream>>>(dw1, 2064, 16,   WF1, 65);
  wprep_kernel<<<9,    256, 0, stream>>>(dw2, 2080, 16,   WF2, 65);
  attn_kernel<<<1, 32, 0, stream>>>(T, S, ln_w, ln_b, wq, bq, wk, bk, wv, bv,
                                    wo, bo, rel, gamma, addvec);
  rdb_gemm_kernel<<<NPIX / 128, 256, 0, stream>>>(XT, WF0, db0, 64, XT, 0);
  rdb_gemm_kernel<<<NPIX / 128, 256, 0, stream>>>(XT, WF1, db1, 65, XT, 16);
  rdb_gemm_kernel<<<NPIX / 128, 256, 0, stream>>>(XT, WF2, db2, 65, XT, 32);
  fused_gemm_kernel<<<dim3(NPIX / 32, 2048 / 128), 256, 0, stream>>>(
      XT, WFf, fb, x, addvec, rdc, outp);
}
